// MoECNBlock_82798379532746
// MI455X (gfx1250) — matrix operand order repaired and checked by reading, not compiled, not run on hardware
//
#include <hip/hip_runtime.h>
#include <hip/hip_bf16.h>
#include <math.h>

// ---------------- problem constants ----------------
constexpr int kN  = 64;
constexpr int kC  = 384;
constexpr int kH  = 32;
constexpr int kW  = 32;
constexpr int kT  = kN * kH * kW;      // 65536 tokens
constexpr int kHD = 4 * kC;            // 1536
constexpr int kE  = 8;
constexpr int kBE = 13107;             // round(2*65536*0.8/8)
constexpr float kLnEps = 1e-6f;

// WMMA tiling
constexpr int kMT  = 32;               // slot rows per block
constexpr int kKT1 = kC / 32;          // 12  k-steps GEMM1
constexpr int kNT1 = kHD / 16;         // 96  n-tiles GEMM1
constexpr int kKT2 = kHD / 32;         // 48  k-steps GEMM2
constexpr int kNT2 = kC / 16;          // 24  n-tiles GEMM2
constexpr int kTilesW1 = kE * kNT1 * kKT1;   // 9216
constexpr int kTilesW2 = kE * kNT2 * kKT2;   // 9216

typedef __bf16 bf16t;
typedef __attribute__((ext_vector_type(16))) __bf16 v16bf;
typedef __attribute__((ext_vector_type(8)))  __bf16 v8bf;
typedef __attribute__((ext_vector_type(8)))  float  v8f;

// ---------------- WMMA helpers ----------------
__device__ __forceinline__ v16bf cat8(v8bf lo, v8bf hi) {
  return __builtin_shufflevector(lo, hi, 0,1,2,3,4,5,6,7,8,9,10,11,12,13,14,15);
}

// A fragment (16x32 bf16) from row-major LDS matrix: lane holds row = lane&15,
// K range [16*half, 16*half+16) contiguous (matches packed B layout).
__device__ __forceinline__ v16bf load_a_frag(const bf16t* base, int ldr,
                                             int row0, int k0, int lane) {
  const int half = lane >> 4;
  const int m    = lane & 15;
  const bf16t* p = base + (size_t)(row0 + m) * ldr + k0 + 16 * half;
  v8bf lo = *(const v8bf*)p;
  v8bf hi = *(const v8bf*)(p + 8);
  return cat8(lo, hi);
}

// B fragment (32x16 bf16) from pre-packed lane-major tile (512 bf16 per tile).
__device__ __forceinline__ v16bf load_b_frag(const bf16t* tile, int lane) {
  const v8bf* p = (const v8bf*)(tile + lane * 16);
  return cat8(p[0], p[1]);
}

__device__ __forceinline__ v8f wmma_bf16(v16bf a, v16bf b, v8f c) {
  return __builtin_amdgcn_wmma_f32_16x16x32_bf16(false, a, false, b,
                                                 (short)0, c, false, false);
}

__device__ __forceinline__ float gelu_exact(float v) {
  return 0.5f * v * (1.0f + erff(v * 0.70710678118654752f));
}

// ---------------- weight packing into WMMA B-fragment layout ----------------
// w: [E][K][Nd] row-major f32 -> wp tiles [e][nt][kt][512] bf16
// value(lane, i) = w[e][kt*32 + 16*(lane>>4) + i][nt*16 + (lane&15)]
__global__ void pack_w_kernel(const float* __restrict__ w, bf16t* __restrict__ wp,
                              int K, int Nd) {
  int gid  = blockIdx.x * 256 + threadIdx.x;
  int lane = gid & 31;
  int tile = gid >> 5;
  int KT = K / 32, NT = Nd / 16;
  if (tile >= kE * NT * KT) return;
  int kt  = tile % KT;
  int rem = tile / KT;
  int nt  = rem % NT;
  int e   = rem / NT;
  int n   = nt * 16 + (lane & 15);
  int kb  = kt * 32 + 16 * (lane >> 4);
  bf16t* dst = wp + (size_t)tile * 512 + lane * 16;
  const float* src = w + ((size_t)e * K + kb) * Nd + n;
#pragma unroll
  for (int i = 0; i < 16; ++i) dst[i] = (bf16t)src[(size_t)i * Nd];
}

// ---------------- depthwise 7x7 conv + bias -> NHWC f32 ----------------
__global__ void dwconv_kernel(const float* __restrict__ x,
                              const float* __restrict__ cw,
                              const float* __restrict__ cb,
                              float* __restrict__ convo) {
  size_t id = (size_t)blockIdx.x * 256 + threadIdx.x;   // t*kC + c
  if (id >= (size_t)kT * kC) return;
  int c = (int)(id % kC);
  size_t t = id / kC;
  int w = (int)(t % kW);
  int h = (int)((t / kW) % kH);
  int n = (int)(t / ((size_t)kW * kH));
  const float* xp = x + ((size_t)n * kC + c) * (kH * kW);
  float acc = cb[c];
  for (int dy = 0; dy < 7; ++dy) {
    int yy = h + dy - 3;
    if (yy < 0 || yy >= kH) continue;
    for (int dx = 0; dx < 7; ++dx) {
      int xx = w + dx - 3;
      if (xx < 0 || xx >= kW) continue;
      acc += xp[yy * kW + xx] * cw[(dy * 7 + dx) * kC + c];
    }
  }
  convo[id] = acc;
}

// ---------------- LayerNorm (in place) + bf16 copy; wave per token ----------------
__global__ void ln_kernel(float* __restrict__ xf, bf16t* __restrict__ xb,
                          const float* __restrict__ g, const float* __restrict__ b) {
  int t    = blockIdx.x * 8 + (threadIdx.x >> 5);
  int lane = threadIdx.x & 31;
  float* row = xf + (size_t)t * kC;
  float v[kC / 32];
  float s = 0.f, ss = 0.f;
#pragma unroll
  for (int i = 0; i < kC / 32; ++i) {
    v[i] = row[lane + 32 * i];
    s += v[i]; ss += v[i] * v[i];
  }
  for (int o = 16; o > 0; o >>= 1) {
    s  += __shfl_xor(s, o, 32);
    ss += __shfl_xor(ss, o, 32);
  }
  float mu  = s / kC;
  float var = ss / kC - mu * mu;
  float rs  = rsqrtf(var + kLnEps);
#pragma unroll
  for (int i = 0; i < kC / 32; ++i) {
    int c = lane + 32 * i;
    float y = (v[i] - mu) * rs * g[c] + b[c];
    row[c] = y;
    xb[(size_t)t * kC + c] = (bf16t)y;
  }
}

// ---------------- router: wave per token ----------------
__global__ void router_kernel(const float* __restrict__ xf, const float* __restrict__ gw,
                              float* __restrict__ pri, int* __restrict__ top0,
                              int* __restrict__ top1, float* __restrict__ wg0,
                              float* __restrict__ wg1, int* __restrict__ cnt0) {
  int t    = blockIdx.x * 8 + (threadIdx.x >> 5);
  int lane = threadIdx.x & 31;
  const float* row = xf + (size_t)t * kC;
  float acc[kE] = {};
#pragma unroll
  for (int i = 0; i < kC / 32; ++i) {
    float xv = row[lane + 32 * i];
#pragma unroll
    for (int e = 0; e < kE; ++e) acc[e] += xv * gw[e * kC + lane + 32 * i];
  }
  for (int o = 16; o > 0; o >>= 1) {
#pragma unroll
    for (int e = 0; e < kE; ++e) acc[e] += __shfl_xor(acc[e], o, 32);
  }
  if (lane == 0) {
    int i0 = 0; float v0 = acc[0];
#pragma unroll
    for (int e = 1; e < kE; ++e) if (acc[e] > v0) { v0 = acc[e]; i0 = e; }
    int i1 = -1; float v1 = -3.4e38f;
#pragma unroll
    for (int e = 0; e < kE; ++e)
      if (e != i0 && acc[e] > v1) { v1 = acc[e]; i1 = e; }
    float sum = 0.f;
#pragma unroll
    for (int e = 0; e < kE; ++e) sum += expf(acc[e] - v0);
    float prio = 1.0f / sum;                // softmax prob of argmax
    float e1 = expf(v1 - v0);               // top-2 weights = softmax([v0,v1])
    float w0 = 1.0f / (1.0f + e1);
    pri[t] = prio; top0[t] = i0; top1[t] = i1;
    wg0[t] = w0;   wg1[t] = e1 / (1.0f + e1);
    atomicAdd(&cnt0[i0], 1);
  }
}

// ---------------- dispatch: rank within (k,expert) bucket ----------------
__global__ void dispatch_kernel(const float* __restrict__ pri,
                                const int* __restrict__ top0, const int* __restrict__ top1,
                                const float* __restrict__ wg0, const float* __restrict__ wg1,
                                const int* __restrict__ cnt0,
                                int* __restrict__ disp_tok, float* __restrict__ disp_gate,
                                int* __restrict__ slotmap) {
  __shared__ float sp[256];
  __shared__ int   se[256];
  const int k = blockIdx.y;
  const int t = blockIdx.x * 256 + threadIdx.x;
  const int*   tope = (k == 0) ? top0 : top1;
  const float* wsel = (k == 0) ? wg0  : wg1;
  const int   e = tope[t];
  const float p = pri[t];
  int rank = 0;
  for (int j0 = 0; j0 < kT; j0 += 256) {
    __syncthreads();
    sp[threadIdx.x] = pri[j0 + threadIdx.x];
    se[threadIdx.x] = tope[j0 + threadIdx.x];
    __syncthreads();
#pragma unroll 4
    for (int j = 0; j < 256; ++j) {
      int tj = j0 + j;
      bool before = (sp[j] > p) || (sp[j] == p && tj < t);
      rank += (se[j] == e && before) ? 1 : 0;
    }
  }
  int pos = (k == 0) ? rank : (cnt0[e] + rank);
  if (pos < kBE) {
    int slot = e * kBE + pos;
    disp_tok[slot]  = t;
    disp_gate[slot] = wsel[t];
    slotmap[t * 2 + k] = slot;
  }
}

// ---------------- fused expert MLP (WMMA bf16) ----------------
__global__ __launch_bounds__(256) void moe_mlp_kernel(
    const bf16t* __restrict__ xb, const bf16t* __restrict__ w1p,
    const bf16t* __restrict__ w2p, const float* __restrict__ b1,
    const float* __restrict__ b2, const int* __restrict__ disp_tok,
    const float* __restrict__ disp_gate, float* __restrict__ ybuf) {
  __shared__ bf16t xa[kMT][kC + 8];        // ~25 KB, padded rows (bank-friendly)
  __shared__ bf16t hid[kMT][kHD + 8];      // ~97 KB
  __shared__ int   stok[kMT];
  __shared__ float sgate[kMT];

  const int e    = blockIdx.y;
  const int m0   = blockIdx.x * kMT;
  const int tid  = threadIdx.x;
  const int lane = tid & 31;
  const int wv   = tid >> 5;               // wave 0..7

  if (tid < kMT) {
    int slot = m0 + tid;
    int tok = 0; float g = 0.f;
    if (slot < kBE) { tok = disp_tok[e * kBE + slot]; g = disp_gate[e * kBE + slot]; }
    stok[tid] = tok; sgate[tid] = g;
  }
  __syncthreads();

  // gather 32 token rows (bf16) into LDS via CDNA5 async DMA:
  // 8 threads/row, 6 x GLOBAL_LOAD_ASYNC_TO_LDS_B128 each (offset applies to
  // both the global and the LDS address; chunks advance identically by 16B).
  {
    int r = tid >> 3, part = tid & 7;
    const bf16t* gsrc = xb + (size_t)stok[r] * kC + part * 48;
    unsigned long long ga = (unsigned long long)(uintptr_t)gsrc;
    // low 32 bits of a generic LDS pointer are the wave-relative LDS offset
    unsigned lds_addr = (unsigned)(uintptr_t)(&xa[r][part * 48]);
#define MOE_ASYNC_LD(OFF)                                                     \
    asm volatile("global_load_async_to_lds_b128 %0, %1, off offset:" #OFF     \
                 :: "v"(lds_addr), "v"(ga) : "memory")
    MOE_ASYNC_LD(0);
    MOE_ASYNC_LD(16);
    MOE_ASYNC_LD(32);
    MOE_ASYNC_LD(48);
    MOE_ASYNC_LD(64);
    MOE_ASYNC_LD(80);
#undef MOE_ASYNC_LD
    asm volatile("s_wait_asynccnt 0x0" ::: "memory");
  }
  __syncthreads();

  // ---- GEMM1: hid = gelu(xa @ w1 + b1), bf16 into LDS ----
  const bf16t* w1e = w1p + (size_t)e * kNT1 * kKT1 * 512;
  for (int nt = wv; nt < kNT1; nt += 8) {
    v8f acc0 = {0.f,0.f,0.f,0.f,0.f,0.f,0.f,0.f};
    v8f acc1 = {0.f,0.f,0.f,0.f,0.f,0.f,0.f,0.f};
    const bf16t* bb = w1e + (size_t)nt * kKT1 * 512;
    for (int kt = 0; kt < kKT1; ++kt) {
      v16bf bf = load_b_frag(bb + kt * 512, lane);
      v16bf a0 = load_a_frag(&xa[0][0], kC + 8, 0,  kt * 32, lane);
      v16bf a1 = load_a_frag(&xa[0][0], kC + 8, 16, kt * 32, lane);
      acc0 = wmma_bf16(a0, bf, acc0);
      acc1 = wmma_bf16(a1, bf, acc1);
    }
    int n = lane & 15, half = lane >> 4;
    float bias = b1[(size_t)e * kHD + nt * 16 + n];
#pragma unroll
    for (int r = 0; r < 8; ++r) {
      int m = r + 8 * half;
      hid[m][nt * 16 + n]      = (bf16t)gelu_exact(acc0[r] + bias);
      hid[m + 16][nt * 16 + n] = (bf16t)gelu_exact(acc1[r] + bias);
    }
  }
  __syncthreads();

  // ---- GEMM2: y = (hid @ w2 + b2) * gate -> ybuf[slot][c] ----
  const bf16t* w2e = w2p + (size_t)e * kNT2 * kKT2 * 512;
  for (int nt = wv; nt < kNT2; nt += 8) {
    v8f acc0 = {0.f,0.f,0.f,0.f,0.f,0.f,0.f,0.f};
    v8f acc1 = {0.f,0.f,0.f,0.f,0.f,0.f,0.f,0.f};
    const bf16t* bb = w2e + (size_t)nt * kKT2 * 512;
    for (int kt = 0; kt < kKT2; ++kt) {
      v16bf bf = load_b_frag(bb + kt * 512, lane);
      v16bf a0 = load_a_frag(&hid[0][0], kHD + 8, 0,  kt * 32, lane);
      v16bf a1 = load_a_frag(&hid[0][0], kHD + 8, 16, kt * 32, lane);
      acc0 = wmma_bf16(a0, bf, acc0);
      acc1 = wmma_bf16(a1, bf, acc1);
    }
    int n = lane & 15, half = lane >> 4;
    float bias = b2[(size_t)e * kC + nt * 16 + n];
#pragma unroll
    for (int r = 0; r < 8; ++r) {
      int m = r + 8 * half;
      if (m0 + m < kBE)
        ybuf[((size_t)e * kBE + m0 + m) * kC + nt * 16 + n] = (acc0[r] + bias) * sgate[m];
      int m2 = m + 16;
      if (m0 + m2 < kBE)
        ybuf[((size_t)e * kBE + m0 + m2) * kC + nt * 16 + n] = (acc1[r] + bias) * sgate[m2];
    }
  }
}

// ---------------- deterministic combine: gather per-token slots ----------------
__global__ void gather_combine_kernel(const float* __restrict__ ybuf,
                                      const int* __restrict__ slotmap,
                                      float* __restrict__ tmp) {
  size_t u = (size_t)blockIdx.x * 256 + threadIdx.x;   // t*kC + c
  if (u >= (size_t)kT * kC) return;
  int c = (int)(u % kC);
  size_t t = u / kC;
  int s0 = slotmap[2 * t], s1 = slotmap[2 * t + 1];
  float v = 0.f;
  if (s0 >= 0) v += ybuf[(size_t)s0 * kC + c];
  if (s1 >= 0) v += ybuf[(size_t)s1 * kC + c];
  tmp[u] = v;
}

// ---------------- residual + layer_scale with NHWC->NCHW LDS transpose ----------------
__global__ void combine_out_kernel(const float* __restrict__ x,
                                   const float* __restrict__ tmp,
                                   const float* __restrict__ ls,
                                   float* __restrict__ out) {
  __shared__ float tile[32][33];
  int bid = blockIdx.x;
  int ct  = bid % (kC / 32);
  int hwt = (bid / (kC / 32)) % ((kH * kW) / 32);
  int n   = bid / ((kC / 32) * ((kH * kW) / 32));
  int c0 = ct * 32, hw0 = hwt * 32;
  int li = threadIdx.x & 31;
  int lj = threadIdx.x >> 5;   // 0..7
#pragma unroll
  for (int s = 0; s < 32; s += 8) {
    int hw = hw0 + lj + s;
    tile[lj + s][li] = tmp[((size_t)n * (kH * kW) + hw) * kC + c0 + li];
  }
  __syncthreads();
#pragma unroll
  for (int s = 0; s < 32; s += 8) {
    int c = c0 + lj + s;
    size_t o = ((size_t)n * kC + c) * (kH * kW) + hw0 + li;
    out[o] = x[o] + ls[c] * tile[li][lj + s];
  }
}

// ---------------- host launcher ----------------
// Workspace requirement: ~335 MB.
extern "C" void kernel_launch(void* const* d_in, const int* in_sizes, int n_in,
                              void* d_out, int out_size, void* d_ws, size_t ws_size,
                              hipStream_t stream) {
  const float* x      = (const float*)d_in[0];
  const float* conv_w = (const float*)d_in[1];
  const float* conv_b = (const float*)d_in[2];
  const float* ln_g   = (const float*)d_in[3];
  const float* ln_b   = (const float*)d_in[4];
  const float* gate_w = (const float*)d_in[5];
  const float* w1     = (const float*)d_in[6];
  const float* b1     = (const float*)d_in[7];
  const float* w2     = (const float*)d_in[8];
  const float* b2     = (const float*)d_in[9];
  const float* lscale = (const float*)d_in[10];
  float* out = (float*)d_out;

  char* base = (char*)d_ws;
  size_t off = 0;
  auto carve = [&](size_t bytes) -> void* {
    void* p = base + off;
    off += (bytes + 255) & ~(size_t)255;
    return p;
  };
  float* xf    = (float*)carve((size_t)kT * kC * 4);          // conv out -> LN (in place)
  bf16t* xbb   = (bf16t*)carve((size_t)kT * kC * 2);          // LN output, bf16
  bf16t* w1p   = (bf16t*)carve((size_t)kTilesW1 * 512 * 2);
  bf16t* w2p   = (bf16t*)carve((size_t)kTilesW2 * 512 * 2);
  float* pri   = (float*)carve((size_t)kT * 4);
  int*   top0  = (int*)carve((size_t)kT * 4);
  int*   top1  = (int*)carve((size_t)kT * 4);
  float* wg0   = (float*)carve((size_t)kT * 4);
  float* wg1   = (float*)carve((size_t)kT * 4);
  int*   cnt0  = (int*)carve(256);
  int*   dtok  = (int*)carve((size_t)kE * kBE * 4);
  float* dgate = (float*)carve((size_t)kE * kBE * 4);
  int*   smap  = (int*)carve((size_t)kT * 2 * 4);
  float* ybuf  = (float*)carve((size_t)kE * kBE * kC * 4);
  float* tmp   = xf;   // alias: xf no longer needed after router

  hipMemsetAsync(cnt0, 0, 256, stream);
  hipMemsetAsync(dtok, 0, (size_t)kE * kBE * 4, stream);
  hipMemsetAsync(dgate, 0, (size_t)kE * kBE * 4, stream);
  hipMemsetAsync(smap, 0xFF, (size_t)kT * 2 * 4, stream);   // -1

  pack_w_kernel<<<(kTilesW1 * 32 + 255) / 256, 256, 0, stream>>>(w1, w1p, kC, kHD);
  pack_w_kernel<<<(kTilesW2 * 32 + 255) / 256, 256, 0, stream>>>(w2, w2p, kHD, kC);

  dwconv_kernel<<<(int)(((size_t)kT * kC) / 256), 256, 0, stream>>>(x, conv_w, conv_b, xf);
  ln_kernel<<<kT / 8, 256, 0, stream>>>(xf, xbb, ln_g, ln_b);
  router_kernel<<<kT / 8, 256, 0, stream>>>(xf, gate_w, pri, top0, top1, wg0, wg1, cnt0);

  dim3 dgrid(kT / 256, 2);
  dispatch_kernel<<<dgrid, 256, 0, stream>>>(pri, top0, top1, wg0, wg1, cnt0,
                                             dtok, dgate, smap);

  dim3 mgrid((kBE + kMT - 1) / kMT, kE);
  moe_mlp_kernel<<<mgrid, 256, 0, stream>>>(xbb, w1p, w2p, b1, b2, dtok, dgate, ybuf);

  gather_combine_kernel<<<(int)(((size_t)kT * kC) / 256), 256, 0, stream>>>(ybuf, smap, tmp);
  combine_out_kernel<<<kN * ((kH * kW) / 32) * (kC / 32), 256, 0, stream>>>(x, tmp, lscale, out);
}
